// Head_67972152426822
// MI455X (gfx1250) — matrix operand order repaired and checked by reading, not compiled, not run on hardware
//
#include <hip/hip_runtime.h>
#include <hip/hip_bf16.h>
#include <math.h>

// Problem sizes (fixed by the reference)
#define NB  512
#define TT  128
#define CC  1024
#define HH  128
#define CHUNK 32            // K-slab per staging iteration == one WMMA k-step
#define LD_S 40             // staging LDS row stride (CHUNK + 8 pad), 80B = 16B multiple
#define LD_M 136            // matrix LDS row stride (128 + 8 pad), 272B = 16B multiple
#define SCALE 11.31370849898476f   // sqrt(128): reference MULTIPLIES by head_size**0.5

typedef __attribute__((ext_vector_type(16))) __bf16 v16bf;
typedef __attribute__((ext_vector_type(8)))  __bf16 v8bf;
typedef __attribute__((ext_vector_type(8)))  float  v8f;

// Force issue order: a group of DS-reads (mask 0x100) before the group of
// matrix ops (mask 0x8) within the current scheduling region, so fragment
// loads stay hoisted in distinct registers instead of being sunk next to
// their WMMA with a full dscnt drain.
#if __has_builtin(__builtin_amdgcn_sched_group_barrier)
#define SCHED_DS_THEN_WMMA(nds, nw)                         \
  do {                                                      \
    __builtin_amdgcn_sched_group_barrier(0x100, (nds), 0);  \
    __builtin_amdgcn_sched_group_barrier(0x008, (nw), 0);   \
  } while (0)
#else
#define SCHED_DS_THEN_WMMA(nds, nw) do {} while (0)
#endif

union FragU { v16bf v; v8bf h[2]; };

// A-matrix fragment (16x32 bf16): lane l<16 holds row M=l, K = k0+0..7 and k0+16..23,
// lane l+16 holds row M=l, K = k0+8..15 and k0+24..31  (two 16B LDS loads)
__device__ __forceinline__ v16bf load_frag_A(const __bf16* base, int row0, int k0,
                                             int ld, int lane) {
  const int hh = lane >> 4, l = lane & 15;
  const __bf16* p = base + (row0 + l) * ld + k0 + 8 * hh;
  FragU u;
  u.h[0] = *(const v8bf*)(p);
  u.h[1] = *(const v8bf*)(p + 16);
  return u.v;
}

// B-matrix fragment (32x16 bf16): lane n<16 holds column N=n, K = k0+0..7 and k0+16..23;
// lane n+16 holds column N=n, K = k0+8..15 and k0+24..31. Source stored [N][K] (two 16B loads).
__device__ __forceinline__ v16bf load_frag_B(const __bf16* base, int col0, int k0,
                                             int ld, int lane) {
  const int hh = lane >> 4, n = lane & 15;
  const __bf16* p = base + (col0 + n) * ld + k0 + 8 * hh;
  FragU u;
  u.h[0] = *(const v8bf*)(p);
  u.h[1] = *(const v8bf*)(p + 16);
  return u.v;
}

__device__ __forceinline__ v8f wmma_bf16(v16bf a, v16bf b, v8f c) {
  return __builtin_amdgcn_wmma_f32_16x16x32_bf16(false, a, false, b,
                                                 (short)0, c, false, false);
}

__global__ __launch_bounds__(512)
void head_attn_kernel(const float* __restrict__ idx,
                      const float* __restrict__ Wk,
                      const float* __restrict__ Wq,
                      const float* __restrict__ Wv,
                      float* __restrict__ out) {
  // ---- LDS carve (all offsets 16B aligned) ---------------------------------
  // sQ:17408  sK:17408  sVt:17408  | stage: sX:5120  sWq/sWk/sWv: 3*5120
  // total 72704 bf16 = 145408 bytes (2 blocks/WGP in 320KB). sP overlays the
  // staging region; softmax cross-wave reduction scratch lives in the tail.
  __shared__ __align__(16) __bf16 smem[72704];
  __bf16* sQ  = smem;
  __bf16* sK  = smem + 17408;
  __bf16* sVt = smem + 34816;
  __bf16* sX  = smem + 52224;
  __bf16* sWq = smem + 57344;
  __bf16* sWk = smem + 62464;
  __bf16* sWv = smem + 67584;
  __bf16* sP  = smem + 52224;                  // reuses staging space
  float*  sRedM = (float*)(smem + 69632);      // 256 floats: partial col max
  float*  sRedS = sRedM + 256;                 // 256 floats: partial col sum

  const int bb   = blockIdx.x;            // batch element
  const int tid  = threadIdx.x;
  const int wave = tid >> 5;              // 16 waves
  const int lane = tid & 31;
  const int w8   = wave & 7;              // primary tile index (row or col band)
  const int wh   = wave >> 3;             // which half of the secondary dim
  const int hh   = lane >> 4;             // lane half
  const int n15  = lane & 15;
  const float* Xg = idx + (size_t)bb * TT * CC;

  // =========================================================================
  // Phase 1: Q/K/V projections, single HBM pass over idx[b].
  // wave (w8, wh) owns row band t = w8*16.. and column tiles wh*4 .. wh*4+3.
  // =========================================================================
  v8f accQ[4] = {}, accK[4] = {}, accV[4] = {};

  for (int c0 = 0; c0 < CC; c0 += CHUNK) {
    // stage X chunk [128 x 32] fp32 -> bf16 (coalesced float4 global loads)
    for (int i = tid; i < TT * (CHUNK / 4); i += 512) {
      const int row = i >> 3, c4 = i & 7;       // CHUNK/4 == 8
      const float4 f = ((const float4*)(Xg + (size_t)row * CC + c0))[c4];
      __bf16* d = sX + row * LD_S + c4 * 4;
      d[0] = (__bf16)f.x; d[1] = (__bf16)f.y; d[2] = (__bf16)f.z; d[3] = (__bf16)f.w;
    }
    // stage three weight chunks, TRANSPOSED: W[c0+ci][h] -> sW*[h][ci]
    for (int i = tid; i < CHUNK * (HH / 4); i += 512) {
      const int ci = i >> 5, h4 = i & 31;       // HH/4 == 32
      const int h = h4 * 4;
      const float4 fq = ((const float4*)(Wq + (size_t)(c0 + ci) * HH))[h4];
      const float4 fk = ((const float4*)(Wk + (size_t)(c0 + ci) * HH))[h4];
      const float4 fv = ((const float4*)(Wv + (size_t)(c0 + ci) * HH))[h4];
      sWq[(h+0)*LD_S + ci] = (__bf16)fq.x; sWq[(h+1)*LD_S + ci] = (__bf16)fq.y;
      sWq[(h+2)*LD_S + ci] = (__bf16)fq.z; sWq[(h+3)*LD_S + ci] = (__bf16)fq.w;
      sWk[(h+0)*LD_S + ci] = (__bf16)fk.x; sWk[(h+1)*LD_S + ci] = (__bf16)fk.y;
      sWk[(h+2)*LD_S + ci] = (__bf16)fk.z; sWk[(h+3)*LD_S + ci] = (__bf16)fk.w;
      sWv[(h+0)*LD_S + ci] = (__bf16)fv.x; sWv[(h+1)*LD_S + ci] = (__bf16)fv.y;
      sWv[(h+2)*LD_S + ci] = (__bf16)fv.z; sWv[(h+3)*LD_S + ci] = (__bf16)fv.w;
    }
    if (c0 + CHUNK < CC)  // warm the next idx slab (global_prefetch_b8)
      __builtin_prefetch(Xg + (size_t)(tid & 127) * CC + c0 + CHUNK, 0, 1);
    __syncthreads();

    // batch-issue: 1 A-frag + 12 B-frags (26 DS reads), then 12 WMMAs
    const v16bf a = load_frag_A(sX, w8 * 16, 0, LD_S, lane);
    v16bf bq[4], bk[4], bv[4];
    #pragma unroll
    for (int j = 0; j < 4; ++j) {
      bq[j] = load_frag_B(sWq, wh * 64 + j * 16, 0, LD_S, lane);
      bk[j] = load_frag_B(sWk, wh * 64 + j * 16, 0, LD_S, lane);
      bv[j] = load_frag_B(sWv, wh * 64 + j * 16, 0, LD_S, lane);
    }
    #pragma unroll
    for (int j = 0; j < 4; ++j) {
      accQ[j] = wmma_bf16(a, bq[j], accQ[j]);
      accK[j] = wmma_bf16(a, bk[j], accK[j]);
      accV[j] = wmma_bf16(a, bv[j], accV[j]);
    }
    SCHED_DS_THEN_WMMA(26, 12);
    __syncthreads();
  }

  // spill Q,K (natural [t][h], scalar) and V transposed ([h][s], packed b128)
  #pragma unroll
  for (int j = 0; j < 4; ++j) {
    const int n = wh * 4 + j;
    v8bf pk;
    #pragma unroll
    for (int r = 0; r < 8; ++r) {
      const int m = w8 * 16 + r + 8 * hh;       // D layout: M = r + 8*half
      sQ[m * LD_M + n * 16 + n15] = (__bf16)accQ[j][r];
      sK[m * LD_M + n * 16 + n15] = (__bf16)accK[j][r];
      pk[r] = (__bf16)accV[j][r];
    }
    // per-lane 8 consecutive M values are contiguous in the transposed layout
    *(v8bf*)(sVt + (n * 16 + n15) * LD_M + w8 * 16 + 8 * hh) = pk;
  }
  __syncthreads();

  // =========================================================================
  // Phase 2: S = Q K^T.  wave (w8, wh): column tile s-band = w8*16, row tiles
  // m = wh*4 .. wh*4+3.  Axis=1 (query-axis) softmax = within-wave reduction
  // + one cross-wave (t-half) merge through LDS.
  // =========================================================================
  v8f sacc[4] = {};
  #pragma unroll
  for (int k = 0; k < HH / 32; ++k) {
    const v16bf bfr = load_frag_B(sK, w8 * 16, k * 32, LD_M, lane);
    v16bf af[4];
    #pragma unroll
    for (int mi = 0; mi < 4; ++mi)
      af[mi] = load_frag_A(sQ, (wh * 4 + mi) * 16, k * 32, LD_M, lane);
    #pragma unroll
    for (int mi = 0; mi < 4; ++mi)
      sacc[mi] = wmma_bf16(af[mi], bfr, sacc[mi]);
  }
  SCHED_DS_THEN_WMMA(40, 16);   // whole unrolled k-loop is one region

  // column-wise causal softmax: column s valid for t in [s, 127]
  const int s = w8 * 16 + n15;
  float mx = -__builtin_inff();
  #pragma unroll
  for (int mi = 0; mi < 4; ++mi) {
    #pragma unroll
    for (int r = 0; r < 8; ++r) {
      const int t = (wh * 4 + mi) * 16 + r + 8 * hh;
      const float v = sacc[mi][r] * SCALE;
      sacc[mi][r] = v;
      if (t >= s) mx = fmaxf(mx, v);
    }
  }
  mx = fmaxf(mx, __shfl_xor(mx, 16));           // merge the two lane halves
  if (hh == 0) sRedM[wh * 128 + s] = mx;        // publish per-t-half partial
  __syncthreads();
  const float gmx = fmaxf(sRedM[s], sRedM[128 + s]);
  float sum = 0.f;
  #pragma unroll
  for (int mi = 0; mi < 4; ++mi) {
    #pragma unroll
    for (int r = 0; r < 8; ++r) {
      const int t = (wh * 4 + mi) * 16 + r + 8 * hh;
      const float p = (t >= s) ? __expf(sacc[mi][r] - gmx) : 0.f;
      sacc[mi][r] = p;
      sum += p;
    }
  }
  sum += __shfl_xor(sum, 16);
  if (hh == 0) sRedS[wh * 128 + s] = sum;
  __syncthreads();
  const float inv = 1.0f / (sRedS[s] + sRedS[128 + s]);
  #pragma unroll
  for (int mi = 0; mi < 4; ++mi) {
    #pragma unroll
    for (int r = 0; r < 8; ++r)
      sP[((wh * 4 + mi) * 16 + r + 8 * hh) * LD_M + s] = (__bf16)(sacc[mi][r] * inv);
  }
  __syncthreads();

  // =========================================================================
  // Phase 3: O = P @ V.  wave (w8, wh): row band t = w8*16, h tiles wh*4..+3
  // =========================================================================
  v8f oacc[4] = {};
  #pragma unroll
  for (int k = 0; k < TT / 32; ++k) {
    const v16bf a = load_frag_A(sP, w8 * 16, k * 32, LD_M, lane);
    v16bf bf_[4];
    #pragma unroll
    for (int ni = 0; ni < 4; ++ni)
      bf_[ni] = load_frag_B(sVt, (wh * 4 + ni) * 16, k * 32, LD_M, lane);
    #pragma unroll
    for (int ni = 0; ni < 4; ++ni)
      oacc[ni] = wmma_bf16(a, bf_[ni], oacc[ni]);
  }
  SCHED_DS_THEN_WMMA(40, 16);

  float* Og = out + (size_t)bb * TT * HH;
  #pragma unroll
  for (int ni = 0; ni < 4; ++ni) {
    #pragma unroll
    for (int r = 0; r < 8; ++r)
      Og[(w8 * 16 + r + 8 * hh) * HH + (wh * 4 + ni) * 16 + n15] = oacc[ni][r];
  }
}

extern "C" void kernel_launch(void* const* d_in, const int* in_sizes, int n_in,
                              void* d_out, int out_size, void* d_ws, size_t ws_size,
                              hipStream_t stream) {
  (void)in_sizes; (void)n_in; (void)out_size; (void)d_ws; (void)ws_size;
  const float* idx = (const float*)d_in[0];
  const float* Wk  = (const float*)d_in[1];
  const float* Wq  = (const float*)d_in[2];
  const float* Wv  = (const float*)d_in[3];
  float* out = (float*)d_out;
  hipLaunchKernelGGL(head_attn_kernel, dim3(NB), dim3(512), 0, stream,
                     idx, Wk, Wq, Wv, out);
}
